// MemoryUnit_46222438040046
// MI455X (gfx1250) — compile-verified, matrix operand order checked by reading, not run on hardware
//
#include <hip/hip_runtime.h>

#define MEM_M   2000
#define MEM_MP  2016                               // padded K for pass B
#define FEA_C   256
#define FMAP_W  32
#define BATCH_N 32
#define NROWS   (BATCH_N * FMAP_W * FMAP_W)        // 32768
#define OUT_ELEMS (BATCH_N * FEA_C * FMAP_W * FMAP_W)  // 8388608
#define SHRINK_V 0.0025f
#define EPS_V    1e-12f
#define ROWS_PER_WG 32
#define THREADS 256
#define MTILES  (MEM_M / 16)    // 125

typedef __bf16 bf16_t;
typedef bf16_t v16bf __attribute__((ext_vector_type(16)));
typedef float  v8f   __attribute__((ext_vector_type(8)));
typedef unsigned int v4u __attribute__((ext_vector_type(4)));
typedef int v8i __attribute__((ext_vector_type(8)));
typedef int v4i __attribute__((ext_vector_type(4)));

struct U128 { unsigned int a, b, c, d; };          // trivially-copyable 16B chunk
union BV16 { v16bf v; bf16_t h[16]; unsigned short u[16]; U128 q[2]; };
union AccV { v8f v; float f[8]; };

// workspace layout (bytes)
#define WS_SQ_OFF   0                                   // 2000 f32
#define WS_WBF_OFF  8192                                // [2000][256] bf16 = 1,024,000
#define WS_WBFT_OFF (8192 + 1024000)                    // [256][2016] bf16 = 1,032,192

// dynamic-LDS layout of the fused kernel (base offset 0: no static __shared__ there)
#define LDS_XS_OFF  0                                   // Xs bf16 [32][256] = 16KB
#define LDS_LG_OFF  (ROWS_PER_WG * FEA_C * 2)           // 16384: Lg f32 [32][2000] = 250KB
#define LDS_XF_OFF  LDS_LG_OFF                          // f32 staging [256][32] (aliases Lg)
#define LDS_PF_OFF  (LDS_LG_OFF + 32768)                // f32 staging [256][32]
#define LDS_TAIL_OFF (LDS_LG_OFF + ROWS_PER_WG * MEM_M * 4)

__device__ __forceinline__ float bf2f(unsigned short u) {
    return __builtin_bit_cast(float, ((unsigned int)u) << 16);
}
__device__ __forceinline__ unsigned short bfbits(float f) {
    return __builtin_bit_cast(unsigned short, (bf16_t)f);
}

#if __has_builtin(__builtin_amdgcn_tensor_load_to_lds)
#define USE_TDM 1
// 2D tile DMA: global (tensor_d0 x tensor_d1, row stride stride0 elems, f32)
//   -> LDS at byte offset lds_off, tile (tile_d0 x tile_d1)
__device__ __forceinline__ void tdm_load_2d_f32(unsigned lds_off, const void* gaddr,
                                                unsigned tensor_d0, unsigned tensor_d1,
                                                unsigned tile_d0, unsigned tile_d1,
                                                unsigned stride0) {
    unsigned long long ga = (unsigned long long)gaddr;
    v4u g0;
    g0[0] = 1u;                                              // count=1 (valid), user mode
    g0[1] = lds_off;                                         // D#.lds_addr
    g0[2] = (unsigned)(ga & 0xFFFFFFFFu);                    // global_addr[31:0]
    g0[3] = (unsigned)((ga >> 32) & 0x1FFFFFFu) | (2u << 30); // global_addr[56:32] | type=2
    v8i g1;
    g1[0] = (int)(2u << 16);                                 // data_size=2 (4 bytes)
    g1[1] = (int)(tensor_d0 << 16);                          // tensor_dim0[15:0] @ bits63:48
    g1[2] = (int)((tensor_d0 >> 16) | (tensor_d1 << 16));    // dim0 hi | dim1 lo
    g1[3] = (int)((tensor_d1 >> 16) | (tile_d0 << 16));      // dim1 hi | tile_dim0
    g1[4] = (int)(tile_d1 & 0xFFFFu);                        // tile_dim1 (tile_dim2=0)
    g1[5] = (int)stride0;                                    // tensor_dim0_stride lo32
    g1[6] = 0;                                               // stride0 hi16 | dim1_stride lo
    g1[7] = 0;
    v4i z4 = {0, 0, 0, 0};
#if defined(__clang_major__) && (__clang_major__ >= 23)
    v8i z8 = {0, 0, 0, 0, 0, 0, 0, 0};
    __builtin_amdgcn_tensor_load_to_lds(g0, g1, z4, z4, z8, 0);
#else
    __builtin_amdgcn_tensor_load_to_lds(g0, g1, z4, z4, 0);
#endif
}
#else
#define USE_TDM 0
#endif

// ---- kernel 0: zero loss slots, sq[i]=||w_i||^2, bf16 weight (row + transposed) ----
__global__ void __launch_bounds__(64)
mu_init_kernel(const float* __restrict__ Wt, float* __restrict__ sq,
               unsigned short* __restrict__ Wbf, unsigned short* __restrict__ WbfT,
               float* __restrict__ out) {
    __shared__ float red[64];
    int m = blockIdx.x;
    int tid = threadIdx.x;
    int c0 = tid * 4;
    float4 w = *(const float4*)(Wt + (size_t)m * FEA_C + c0);
    unsigned short b0 = bfbits(w.x), b1 = bfbits(w.y), b2 = bfbits(w.z), b3 = bfbits(w.w);
    unsigned int lo = (unsigned int)b0 | ((unsigned int)b1 << 16);
    unsigned int hi = (unsigned int)b2 | ((unsigned int)b3 << 16);
    *(uint2*)(Wbf + (size_t)m * FEA_C + c0) = make_uint2(lo, hi);
    WbfT[(size_t)(c0 + 0) * MEM_MP + m] = b0;
    WbfT[(size_t)(c0 + 1) * MEM_MP + m] = b1;
    WbfT[(size_t)(c0 + 2) * MEM_MP + m] = b2;
    WbfT[(size_t)(c0 + 3) * MEM_MP + m] = b3;
    red[tid] = w.x * w.x + w.y * w.y + w.z * w.z + w.w * w.w;
    __syncthreads();
    if (tid == 0) {
        float s = 0.f;
        for (int j = 0; j < 64; ++j) s += red[j];
        sq[m] = s;
    }
    if (m == 0) {
        if (tid < 2) out[OUT_ELEMS + tid] = 0.f;
        for (int i = tid; i < FEA_C * (MEM_MP - MEM_M); i += 64) {
            int c = i / (MEM_MP - MEM_M);
            int mm = MEM_M + (i % (MEM_MP - MEM_M));
            WbfT[(size_t)c * MEM_MP + mm] = 0;   // zero K padding (avoid NaN*0)
        }
    }
}

// ---------------- kernel 1: fused logits -> softmax/shrink -> out ----------------
__global__ void __launch_bounds__(THREADS)
mu_fused_kernel(const float* __restrict__ X, const float* __restrict__ Wt,
                const float* __restrict__ Pos,
                const unsigned short* __restrict__ Wbf,
                const unsigned short* __restrict__ WbfT,
                float* __restrict__ Out) {
    extern __shared__ char smem[];
    unsigned short* Xs = (unsigned short*)(smem + LDS_XS_OFF);   // [32][256] bf16
    float* Lg = (float*)(smem + LDS_LG_OFF);                     // [32][2000] f32
    char* tail = smem + LDS_TAIL_OFF;
    float* redF     = (float*)tail;                 // [256]
    int*   redI     = (int*)(tail + 1024);          // [256]
    float* rowMax   = (float*)(tail + 2048);        // [32]
    int*   rowArg   = (int*)(tail + 2176);          // [32]
    float* rowInvZ  = (float*)(tail + 2304);        // [32]
    float* rowInvS1 = (float*)(tail + 2432);        // [32]

    const int t = threadIdx.x;
    const int lane = t & 31;
    const int wid = t >> 5;
    const int hi = (lane >> 4) << 3;        // 0 or 8 (K-half select)
    const int rowBase = blockIdx.x * ROWS_PER_WG;
    const int bIdx = rowBase >> 10;
    const int hw0  = rowBase & 1023;

    // ---- phase 0: stage X + pos tiles, convert to bf16 [n][c] in LDS ----
#if USE_TDM
    {
        // TDM: 2D tile (32 fast elems x 256 channels), aliases the Lg region
        float* Xf   = (float*)(smem + LDS_XF_OFF);   // [256][32] f32
        float* PosF = (float*)(smem + LDS_PF_OFF);   // [256][32] f32
        if (wid == 0) {
            const float* xg = X + (size_t)bIdx * (FEA_C * 1024) + hw0;
            const float* pg = Pos + hw0;
            tdm_load_2d_f32(LDS_XF_OFF, xg, 1024, FEA_C, ROWS_PER_WG, FEA_C, 1024);
            tdm_load_2d_f32(LDS_PF_OFF, pg, 1024, FEA_C, ROWS_PER_WG, FEA_C, 1024);
            __builtin_amdgcn_s_wait_tensorcnt(0);
        }
        __syncthreads();
        int nl = t & 31;
        int cg = t >> 5;
#pragma unroll 4
        for (int ci = 0; ci < FEA_C / 8; ++ci) {
            int c = cg + ci * 8;
            float v = Xf[c * ROWS_PER_WG + nl] + PosF[c * ROWS_PER_WG + nl];
            Xs[nl * FEA_C + c] = bfbits(v);
        }
    }
#else
    {
        int r = t & 31;
        int cg = t >> 5;
        const float* xb = X + (size_t)bIdx * (FEA_C * 1024) + hw0 + r;
        const float* pb = Pos + hw0 + r;
#pragma unroll 4
        for (int ci = 0; ci < FEA_C / 8; ++ci) {
            int c = cg + ci * 8;
            float v = xb[(size_t)c * 1024] + pb[(size_t)c * 1024];
            Xs[r * FEA_C + c] = bfbits(v);
        }
    }
#endif
    __syncthreads();

    // ---- phase 1: logits = X @ W^T (bf16 WMMA), stored to LDS ----
    const int rowTile = wid & 1;
    const int mGroup = wid >> 1;
    BV16 Aop[8];
    {
        int mrow = rowTile * 16 + (lane & 15);
        for (int kc = 0; kc < 8; ++kc) {
            const unsigned short* xp = &Xs[mrow * FEA_C + kc * 32 + hi];
            Aop[kc].q[0] = *(const U128*)(xp);
            Aop[kc].q[1] = *(const U128*)(xp + 16);
        }
    }
    for (int mTile = mGroup; mTile < MTILES; mTile += 4) {
        AccV acc;
        acc.v = (v8f){0.f, 0.f, 0.f, 0.f, 0.f, 0.f, 0.f, 0.f};
        const unsigned short* wb = Wbf + (size_t)(mTile * 16 + (lane & 15)) * FEA_C;
#pragma unroll
        for (int kc = 0; kc < 8; ++kc) {
            int kb = kc * 32 + hi;
            BV16 Bop;
            Bop.q[0] = *(const U128*)(wb + kb);
            Bop.q[1] = *(const U128*)(wb + kb + 16);
            acc.v = __builtin_amdgcn_wmma_f32_16x16x32_bf16(
                false, Aop[kc].v, false, Bop.v, (short)0, acc.v, false, false);
        }
        int nbase = rowTile * 16 + hi;
        int col = mTile * 16 + (lane & 15);
#pragma unroll
        for (int v = 0; v < 8; ++v)
            Lg[(nbase + v) * MEM_M + col] = acc.f[v];
    }
    __syncthreads();

    // ---- phase 2: row reductions; overwrite logits with shrunk att in place ----
    {
        int r = t >> 3;
        int sg = t & 7;
        float* lr = &Lg[r * MEM_M];
        int m0 = sg * 250;
        float mx = -3.4e38f; int amx = m0;
        for (int m = m0; m < m0 + 250; ++m) {
            float v = lr[m];
            if (v > mx) { mx = v; amx = m; }
        }
        redF[t] = mx; redI[t] = amx;
        __syncthreads();
        if (sg == 0) {
            float bm = redF[r * 8]; int ba = redI[r * 8];
            for (int j = 1; j < 8; ++j) {
                float v = redF[r * 8 + j];
                if (v > bm) { bm = v; ba = redI[r * 8 + j]; }
            }
            rowMax[r] = bm; rowArg[r] = ba;
        }
        __syncthreads();
        float bm = rowMax[r];
        float se = 0.f;
        for (int m = m0; m < m0 + 250; ++m) se += __expf(lr[m] - bm);
        redF[t] = se;
        __syncthreads();
        if (sg == 0) {
            float z = 0.f;
            for (int j = 0; j < 8; ++j) z += redF[r * 8 + j];
            rowInvZ[r] = 1.0f / fmaxf(z, 1e-30f);
        }
        __syncthreads();
        float iz = rowInvZ[r];
        float s1 = 0.f;
        for (int m = m0; m < m0 + 250; ++m) {
            float a = __expf(lr[m] - bm) * iz;
            float d = a - SHRINK_V;
            float ap = (d > 0.f ? d : 0.f) * a / (fabsf(d) + EPS_V);
            lr[m] = ap;                       // in-place: logits -> shrunk att
            s1 += ap;
        }
        redF[t] = s1;
        __syncthreads();
        if (sg == 0) {
            float s = 0.f;
            for (int j = 0; j < 8; ++j) s += redF[r * 8 + j];
            rowInvS1[r] = 1.0f / fmaxf(s, EPS_V);
        }
        // compact loss: ||x_r - w[argmax]||^2 partials
        const float* wrow = Wt + (size_t)rowArg[r] * FEA_C;
        float cl = 0.f;
        for (int c = sg * 32; c < sg * 32 + 32; ++c) {
            float d = bf2f(Xs[r * FEA_C + c]) - wrow[c];
            cl += d * d;
        }
        __syncthreads();
        redF[t] = cl;
        __syncthreads();
        if (sg == 0) {
            float s = 0.f;
            for (int j = 0; j < 8; ++j) s += redF[r * 8 + j];
            atomicAdd(Out + OUT_ELEMS, s * (1.0f / ((float)NROWS * (float)FEA_C)));
        }
    }
    __syncthreads();

    // ---- phase 3: out^T = W^T x att^T (bf16 WMMA); no barriers, no staging ----
    const int nTile = wid & 1;
    const int cSel = wid >> 1;               // cTiles: cSel, cSel+4, cSel+8, cSel+12
    const int nloc = nTile * 16 + (lane & 15);
    const float* lr = &Lg[nloc * MEM_M];
    AccV acc[4];
#pragma unroll
    for (int q = 0; q < 4; ++q)
        acc[q].v = (v8f){0.f, 0.f, 0.f, 0.f, 0.f, 0.f, 0.f, 0.f};

    for (int kc = 0; kc < 63; ++kc) {          // 63*32 = 2016 (padded tail)
        int kb = kc * 32 + hi;
        // B operand: att^T chunk from LDS (already shrunk, f32 -> bf16)
        BV16 Bop;
        if (kc < 62) {
            float4 a0 = *(const float4*)(lr + kb);
            float4 a1 = *(const float4*)(lr + kb + 4);
            float4 a2 = *(const float4*)(lr + kb + 16);
            float4 a3 = *(const float4*)(lr + kb + 20);
            Bop.h[0]  = (bf16_t)a0.x;  Bop.h[1]  = (bf16_t)a0.y;
            Bop.h[2]  = (bf16_t)a0.z;  Bop.h[3]  = (bf16_t)a0.w;
            Bop.h[4]  = (bf16_t)a1.x;  Bop.h[5]  = (bf16_t)a1.y;
            Bop.h[6]  = (bf16_t)a1.z;  Bop.h[7]  = (bf16_t)a1.w;
            Bop.h[8]  = (bf16_t)a2.x;  Bop.h[9]  = (bf16_t)a2.y;
            Bop.h[10] = (bf16_t)a2.z;  Bop.h[11] = (bf16_t)a2.w;
            Bop.h[12] = (bf16_t)a3.x;  Bop.h[13] = (bf16_t)a3.y;
            Bop.h[14] = (bf16_t)a3.z;  Bop.h[15] = (bf16_t)a3.w;
        } else {
#pragma unroll
            for (int j = 0; j < 8; ++j) {
                int m0 = kb + j, m1 = kb + 16 + j;
                Bop.h[j]     = (bf16_t)(m0 < MEM_M ? lr[m0] : 0.f);
                Bop.h[j + 8] = (bf16_t)(m1 < MEM_M ? lr[m1] : 0.f);
            }
        }
        // A operands: transposed bf16 weights direct from global (L2-resident)
#pragma unroll
        for (int q = 0; q < 4; ++q) {
            int c = (cSel + q * 4) * 16 + (lane & 15);
            const unsigned short* wtp = WbfT + (size_t)c * MEM_MP + kb;
            BV16 Aw;
            Aw.q[0] = *(const U128*)(wtp);
            Aw.q[1] = *(const U128*)(wtp + 16);
            acc[q].v = __builtin_amdgcn_wmma_f32_16x16x32_bf16(
                false, Aw.v, false, Bop.v, (short)0, acc[q].v, false, false);
        }
    }
    // store: lane axis = n (contiguous), VGPR axis = c
    {
        int n = rowBase + nloc;
        float is1 = rowInvS1[nloc];
        int b = n >> 10, hw = n & 1023;
        float* ob = Out + (size_t)b * (FEA_C * 1024) + hw;
#pragma unroll
        for (int q = 0; q < 4; ++q) {
            int cb = (cSel + q * 4) * 16 + hi;
#pragma unroll
            for (int v = 0; v < 8; ++v)
                ob[(size_t)(cb + v) * 1024] = acc[q].f[v] * is1;
        }
    }
}

// ---------------- kernel 2: distance loss (gram matrix hinge, WMMA) ----------------
__global__ void __launch_bounds__(32)
mu_dist_kernel(const unsigned short* __restrict__ Wbf, const float* __restrict__ sq,
               float* __restrict__ out) {
    int iT = blockIdx.x, jT = blockIdx.y;
    if (jT < iT) return;     // uniform over the single wave
    int lane = threadIdx.x;
    int hi = (lane >> 4) << 3;
    const unsigned short* wa = Wbf + (size_t)(iT * 16 + (lane & 15)) * FEA_C;
    const unsigned short* wb = Wbf + (size_t)(jT * 16 + (lane & 15)) * FEA_C;
    AccV acc;
    acc.v = (v8f){0.f, 0.f, 0.f, 0.f, 0.f, 0.f, 0.f, 0.f};
#pragma unroll
    for (int kc = 0; kc < 8; ++kc) {
        int kb = kc * 32 + hi;
        BV16 A, B;
        A.q[0] = *(const U128*)(wa + kb);
        A.q[1] = *(const U128*)(wa + kb + 16);
        B.q[0] = *(const U128*)(wb + kb);
        B.q[1] = *(const U128*)(wb + kb + 16);
        acc.v = __builtin_amdgcn_wmma_f32_16x16x32_bf16(
            false, A.v, false, B.v, (short)0, acc.v, false, false);
    }
    int j = jT * 16 + (lane & 15);
    float sqj = sq[j];
    float part = 0.f;
#pragma unroll
    for (int v = 0; v < 8; ++v) {
        int i = iT * 16 + v + hi;
        if (j > i) {
            float d2 = sq[i] + sqj - 2.f * acc.f[v];
            float dist = 1.f - d2;
            part += dist > 0.f ? dist : 0.f;
        }
    }
    __shared__ float red[32];
    red[lane] = part;
    __syncthreads();
    if (lane == 0) {
        float s = 0.f;
        for (int k = 0; k < 32; ++k) s += red[k];
        atomicAdd(out + OUT_ELEMS + 1,
                  s * (2.0f / ((float)MEM_M * (float)(MEM_M - 1))));
    }
}

extern "C" void kernel_launch(void* const* d_in, const int* in_sizes, int n_in,
                              void* d_out, int out_size, void* d_ws, size_t ws_size,
                              hipStream_t stream) {
    (void)in_sizes; (void)n_in; (void)out_size; (void)ws_size;
    const float* X   = (const float*)d_in[0];   // [32,256,32,32]
    const float* Wt  = (const float*)d_in[2];   // [2000,256]
    const float* Pos = (const float*)d_in[3];   // [1,256,32,32]
    float* out = (float*)d_out;                 // [out tensor | compact | distance]
    char* ws = (char*)d_ws;
    float* sq = (float*)(ws + WS_SQ_OFF);
    unsigned short* Wbf  = (unsigned short*)(ws + WS_WBF_OFF);
    unsigned short* WbfT = (unsigned short*)(ws + WS_WBFT_OFF);

    mu_init_kernel<<<MEM_M, 64, 0, stream>>>(Wt, sq, Wbf, WbfT, out);

    size_t smem = (size_t)ROWS_PER_WG * FEA_C * 2     // Xs bf16
                + (size_t)ROWS_PER_WG * MEM_M * 4     // logits/att f32 (aliases TDM staging)
                + 2560;                               // reductions
    mu_fused_kernel<<<NROWS / ROWS_PER_WG, THREADS, smem, stream>>>(X, Wt, Pos, Wbf, WbfT, out);

    dim3 dgrid(MTILES, MTILES);
    mu_dist_kernel<<<dgrid, 32, 0, stream>>>(Wbf, sq, out);
}
